// TimeStaticGCNGAT_7885559955676
// MI455X (gfx1250) — compile-verified, matrix operand order checked by reading, not run on hardware
//
#include <hip/hip_runtime.h>
#include <hip/hip_bf16.h>

#define DEVINL __device__ __forceinline__

typedef __attribute__((ext_vector_type(16))) __bf16 v16bf;
typedef __attribute__((ext_vector_type(8)))  __bf16 v8bf;
typedef __attribute__((ext_vector_type(8)))  float  v8f;

constexpr int C_IN = 128, HID = 64, HEADS = 3, C_OUT = 64, HF = HEADS * C_OUT; // 192
constexpr float NEG_SLOPE = 0.2f;

DEVINL __bf16 f2bf(float f) {
  unsigned u = __float_as_uint(f);
  unsigned r = (u + 0x7FFFu + ((u >> 16) & 1u)) >> 16;   // round-to-nearest-even
  unsigned short s = (unsigned short)r;
  return __builtin_bit_cast(__bf16, s);
}

DEVINL void atomicMaxF(float* addr, float v) {
  // sign-punned float atomic max (works with -inf init)
  if (v >= 0.f) atomicMax((int*)addr, __float_as_int(v));
  else          atomicMin((unsigned int*)addr, __float_as_uint(v));
}

// Build a 16x32 (A) / 32x16 (pre-transposed B) bf16 WMMA fragment.
// Per CDNA5 ISA: lane half selects K-offset +8; vector elems 0..7 -> K base..base+7,
// elems 8..15 -> K base+16..base+23. Two contiguous 16B loads.
DEVINL v16bf load_frag(const __bf16* p) {
  v8bf lo = *(const v8bf*)(p);
  v8bf hi = *(const v8bf*)(p + 16);
  v16bf r;
#pragma unroll
  for (int i = 0; i < 8; ++i) { r[i] = lo[i]; r[i + 8] = hi[i]; }
  return r;
}

// ---------------- init: deg=1 (self-loop), x=0, amax=-inf, denom=0, partial=0 ----
__global__ void k_init(float* deg, float* x, float* amax, float* denom, float* part, int N) {
  int i = blockIdx.x * blockDim.x + threadIdx.x;
  int st = gridDim.x * blockDim.x;
  for (int j = i; j < N; j += st) deg[j] = 1.0f;
  for (int j = i; j < N * HID; j += st) x[j] = 0.0f;
  for (int j = i; j < N * HEADS; j += st) { amax[j] = __uint_as_float(0xFF800000u); denom[j] = 0.0f; }
  for (int j = i; j < HF; j += st) part[j] = 0.0f;
}

// ---------------- degree accumulation (real edges only; self-loops pre-seeded) ---
__global__ void k_deg(const int* __restrict__ dst, const float* __restrict__ ew,
                      float* __restrict__ deg, int E) {
  int e = blockIdx.x * blockDim.x + threadIdx.x;
  if (e < E) atomicAdd(&deg[dst[e]], ew[e]);
}

__global__ void k_rsqrt(float* dis, int N) {
  int i = blockIdx.x * blockDim.x + threadIdx.x;
  if (i < N) { float d = dis[i]; dis[i] = d > 0.f ? rsqrtf(d) : 0.f; }
}

// ---------------- fp32 -> bf16 conversions; W1/W2 pre-transposed (N-major, K-contig)
__global__ void k_convert(const float* __restrict__ X, const float* __restrict__ W1,
                          const float* __restrict__ W2, __bf16* Xbf, __bf16* W1t,
                          __bf16* W2t, int N) {
  int i = blockIdx.x * blockDim.x + threadIdx.x;
  int st = gridDim.x * blockDim.x;
  for (int j = i; j < N * C_IN; j += st) Xbf[j] = f2bf(X[j]);
  for (int j = i; j < HID * C_IN; j += st) { int n = j / C_IN, k = j % C_IN; W1t[j] = f2bf(W1[k * HID + n]); }
  for (int j = i; j < HF * HID; j += st)  { int n = j / HID,  k = j % HID;  W2t[j] = f2bf(W2[k * HF + n]); }
}

// ---------------- GEMM1: xw[M x 64] = Xbf[M x 128] @ W1 (W1t is 64 x 128) --------
// 128 threads = 4 waves; wave w owns N-tile w; block owns 16 rows.
__global__ void k_gemm1(const __bf16* __restrict__ A, const __bf16* __restrict__ Bt,
                        float* __restrict__ C, int M) {
  int wave = threadIdx.x >> 5;
  int lane = threadIdx.x & 31;
  int l16 = lane & 15, half = lane >> 4;
  int r0 = blockIdx.x * 16;
  int row = r0 + l16; if (row >= M) row = M - 1;
  int col = wave * 16 + l16;
  const __bf16* ap = A  + (size_t)row * C_IN + half * 8;
  const __bf16* bp = Bt + (size_t)col * C_IN + half * 8;
  v8f acc = {};
#pragma unroll
  for (int kt = 0; kt < C_IN / 32; ++kt) {
    v16bf a = load_frag(ap + kt * 32);
    v16bf b = load_frag(bp + kt * 32);
    acc = __builtin_amdgcn_wmma_f32_16x16x32_bf16(false, a, false, b, (short)0, acc, false, false);
  }
  float* cp = C + (size_t)(r0 + half * 8) * HID + wave * 16 + l16;
  if (r0 + 16 <= M) {               // full tile: branch-free store clause
#pragma unroll
    for (int r = 0; r < 8; ++r) cp[(size_t)r * HID] = acc[r];
  } else {                          // ragged tail (not hit when M % 16 == 0)
#pragma unroll
    for (int r = 0; r < 8; ++r) {
      int rr = r0 + r + half * 8;
      if (rr < M) C[(size_t)rr * HID + wave * 16 + l16] = acc[r];
    }
  }
}

// ---------------- GCN scatter: x[dst] += xw[src] * norm (64 threads / edge) ------
__global__ void k_scatter1(const int* __restrict__ src, const int* __restrict__ dst,
                           const float* __restrict__ ew, const float* __restrict__ dis,
                           const float* __restrict__ xw, float* __restrict__ x,
                           int E, int Et) {
  int grp = blockIdx.x * 4 + (threadIdx.x >> 6);
  int t = threadIdx.x & 63;
  if (grp >= Et) return;
  int s, d; float w;
  if (grp < E) { s = src[grp]; d = dst[grp]; w = ew[grp]; }
  else         { s = d = grp - E; w = 1.0f; }
  float norm = dis[s] * w * dis[d];
  atomicAdd(&x[(size_t)d * HID + t], xw[(size_t)s * HID + t] * norm);
}

__global__ void k_relu_bias(const float* __restrict__ x, const float* __restrict__ b1,
                            __bf16* __restrict__ xbf, int N) {
  int i = blockIdx.x * blockDim.x + threadIdx.x;
  int st = gridDim.x * blockDim.x;
  for (int j = i; j < N * HID; j += st) {
    float v = x[j] + b1[j & (HID - 1)];
    xbf[j] = f2bf(v > 0.f ? v : 0.f);
  }
}

// ---------------- GEMM2: g[M x 192] = xbf[M x 64] @ W2 (W2t is 192 x 64) ---------
__global__ void k_gemm2(const __bf16* __restrict__ A, const __bf16* __restrict__ Bt,
                        float* __restrict__ C, int M) {
  int wave = threadIdx.x >> 5;
  int lane = threadIdx.x & 31;
  int l16 = lane & 15, half = lane >> 4;
  int r0 = blockIdx.x * 16;
  int row = r0 + l16; if (row >= M) row = M - 1;
  bool full = (r0 + 16 <= M);
  v16bf a0 = load_frag(A + (size_t)row * HID + half * 8);
  v16bf a1 = load_frag(A + (size_t)row * HID + half * 8 + 32);
#pragma unroll
  for (int ct = 0; ct < 3; ++ct) {
    int cb = (wave * 3 + ct) * 16;
    const __bf16* bp = Bt + (size_t)(cb + l16) * HID + half * 8;
    v16bf b0 = load_frag(bp);
    v16bf b1 = load_frag(bp + 32);
    v8f acc = {};
    acc = __builtin_amdgcn_wmma_f32_16x16x32_bf16(false, a0, false, b0, (short)0, acc, false, false);
    acc = __builtin_amdgcn_wmma_f32_16x16x32_bf16(false, a1, false, b1, (short)0, acc, false, false);
    float* cp = C + (size_t)(r0 + half * 8) * HF + cb + l16;
    if (full) {
#pragma unroll
      for (int r = 0; r < 8; ++r) cp[(size_t)r * HF] = acc[r];
    } else {
#pragma unroll
      for (int r = 0; r < 8; ++r) {
        int rr = r0 + r + half * 8;
        if (rr < M) C[(size_t)rr * HF + cb + l16] = acc[r];
      }
    }
  }
}

// ---------------- per-node attention projections (1 wave / node) -----------------
__global__ void k_att(const float* __restrict__ g, const float* __restrict__ as,
                      const float* __restrict__ ad, float* __restrict__ a_src,
                      float* __restrict__ a_dst, int N) {
  int n = blockIdx.x * 8 + (threadIdx.x >> 5);
  int lane = threadIdx.x & 31;
  if (n >= N) return;
  const float* gp = g + (size_t)n * HF;
  float ss[HEADS], dd[HEADS];
#pragma unroll
  for (int h = 0; h < HEADS; ++h) {
    float g0 = gp[h * 64 + lane], g1 = gp[h * 64 + lane + 32];
    ss[h] = g0 * as[h * 64 + lane] + g1 * as[h * 64 + lane + 32];
    dd[h] = g0 * ad[h * 64 + lane] + g1 * ad[h * 64 + lane + 32];
  }
#pragma unroll
  for (int h = 0; h < HEADS; ++h)
    for (int off = 16; off; off >>= 1) {
      ss[h] += __shfl_xor(ss[h], off, 32);
      dd[h] += __shfl_xor(dd[h], off, 32);
    }
  if (lane == 0) {
#pragma unroll
    for (int h = 0; h < HEADS; ++h) {
      a_src[(size_t)n * HEADS + h] = ss[h];
      a_dst[(size_t)n * HEADS + h] = dd[h];
    }
  }
}

// ---------------- per-edge alpha + segment max -----------------------------------
__global__ void k_alpha(const int* __restrict__ src, const int* __restrict__ dst,
                        const float* __restrict__ a_src, const float* __restrict__ a_dst,
                        float* __restrict__ alpha, float* __restrict__ amax, int E, int Et) {
  int e = blockIdx.x * blockDim.x + threadIdx.x;
  if (e >= Et) return;
  int s, d;
  if (e < E) { s = src[e]; d = dst[e]; } else { s = d = e - E; }
#pragma unroll
  for (int h = 0; h < HEADS; ++h) {
    float al = a_src[(size_t)s * HEADS + h] + a_dst[(size_t)d * HEADS + h];
    al = al > 0.f ? al : NEG_SLOPE * al;
    alpha[(size_t)e * HEADS + h] = al;
    atomicMaxF(&amax[(size_t)d * HEADS + h], al);
  }
}

// ---------------- exp + segment sum (alpha buffer reused for ex) -----------------
__global__ void k_exp(const int* __restrict__ dst, float* __restrict__ alpha,
                      const float* __restrict__ amax, float* __restrict__ den, int E, int Et) {
  int e = blockIdx.x * blockDim.x + threadIdx.x;
  if (e >= Et) return;
  int d = (e < E) ? dst[e] : e - E;
#pragma unroll
  for (int h = 0; h < HEADS; ++h) {
    float ex = __expf(alpha[(size_t)e * HEADS + h] - amax[(size_t)d * HEADS + h]);
    alpha[(size_t)e * HEADS + h] = ex;
    atomicAdd(&den[(size_t)d * HEADS + h], ex);
  }
}

// ---------------- global aggregation: partial += g[src] * attn (mean-only output)
__global__ void k_aggregate(const int* __restrict__ src, const int* __restrict__ dst,
                            const float* __restrict__ ex, const float* __restrict__ den,
                            const float* __restrict__ g, float* __restrict__ part,
                            int E, int Et) {
  __shared__ float sacc[HF];
  for (int j = threadIdx.x; j < HF; j += blockDim.x) sacc[j] = 0.f;
  __syncthreads();
  int t = threadIdx.x & 63;
  float acc0 = 0.f, acc1 = 0.f, acc2 = 0.f;
  for (int e = blockIdx.x * 4 + (threadIdx.x >> 6); e < Et; e += gridDim.x * 4) {
    int s, d;
    if (e < E) { s = src[e]; d = dst[e]; } else { s = d = e - E; }
    float a0 = ex[(size_t)e * HEADS + 0] / (den[(size_t)d * HEADS + 0] + 1e-16f);
    float a1 = ex[(size_t)e * HEADS + 1] / (den[(size_t)d * HEADS + 1] + 1e-16f);
    float a2 = ex[(size_t)e * HEADS + 2] / (den[(size_t)d * HEADS + 2] + 1e-16f);
    const float* gp = g + (size_t)s * HF;
    acc0 += gp[t]        * a0;
    acc1 += gp[64  + t]  * a1;
    acc2 += gp[128 + t]  * a2;
  }
  atomicAdd(&sacc[t],        acc0);
  atomicAdd(&sacc[64  + t],  acc1);
  atomicAdd(&sacc[128 + t],  acc2);
  __syncthreads();
  for (int j = threadIdx.x; j < HF; j += blockDim.x) atomicAdd(&part[j], sacc[j]);
}

__global__ void k_final(const float* __restrict__ part, const float* __restrict__ b2,
                        float* __restrict__ out, int N) {
  int j = threadIdx.x;
  if (j < HF) out[j] = part[j] / (float)N + b2[j];
}

extern "C" void kernel_launch(void* const* d_in, const int* in_sizes, int n_in,
                              void* d_out, int out_size, void* d_ws, size_t ws_size,
                              hipStream_t stream) {
  const float* nf      = (const float*)d_in[0];
  const int*   ei      = (const int*)d_in[1];
  const float* ea      = (const float*)d_in[2];
  const float* W1      = (const float*)d_in[3];
  const float* b1      = (const float*)d_in[4];
  const float* W2      = (const float*)d_in[5];
  const float* b2      = (const float*)d_in[6];
  const float* att_src = (const float*)d_in[7];
  const float* att_dst = (const float*)d_in[8];

  const int N  = in_sizes[0] / C_IN;
  const int E  = in_sizes[1] / 2;
  const int Et = E + N;
  const int* src = ei;
  const int* dst = ei + E;

  // workspace carving (units: floats, 1KB-aligned blocks)
  size_t off = 0;
  auto alloc = [&](size_t n) { size_t o = off; off += (n + 255) & ~(size_t)255; return o; };
  float* base  = (float*)d_ws;
  float* dis   = base + alloc((size_t)N);             // deg -> rsqrt in place
  float* xw    = base + alloc((size_t)N * HID);
  float* x     = base + alloc((size_t)N * HID);
  float* g     = base + alloc((size_t)N * HF);
  float* asrc  = base + alloc((size_t)N * HEADS);
  float* adst  = base + alloc((size_t)N * HEADS);
  float* amax  = base + alloc((size_t)N * HEADS);
  float* den   = base + alloc((size_t)N * HEADS);
  float* alpha = base + alloc((size_t)Et * HEADS);    // reused: alpha -> ex
  float* part  = base + alloc((size_t)HF);
  __bf16* Xbf  = (__bf16*)(base + alloc((size_t)N * C_IN / 2));
  __bf16* W1t  = (__bf16*)(base + alloc((size_t)HID * C_IN / 2));
  __bf16* xbf  = (__bf16*)(base + alloc((size_t)N * HID / 2));
  __bf16* W2t  = (__bf16*)(base + alloc((size_t)HF * HID / 2));

  k_init     <<<2048, 256, 0, stream>>>(dis, x, amax, den, part, N);
  k_deg      <<<(E + 255) / 256, 256, 0, stream>>>(dst, ea, dis, E);
  k_rsqrt    <<<(N + 255) / 256, 256, 0, stream>>>(dis, N);
  k_convert  <<<2048, 256, 0, stream>>>(nf, W1, W2, Xbf, W1t, W2t, N);
  k_gemm1    <<<(N + 15) / 16, 128, 0, stream>>>(Xbf, W1t, xw, N);
  k_scatter1 <<<(Et + 3) / 4, 256, 0, stream>>>(src, dst, ea, dis, xw, x, E, Et);
  k_relu_bias<<<2048, 256, 0, stream>>>(x, b1, xbf, N);
  k_gemm2    <<<(N + 15) / 16, 128, 0, stream>>>(xbf, W2t, g, N);
  k_att      <<<(N + 7) / 8, 256, 0, stream>>>(g, att_src, att_dst, asrc, adst, N);
  k_alpha    <<<(Et + 255) / 256, 256, 0, stream>>>(src, dst, asrc, adst, alpha, amax, E, Et);
  k_exp      <<<(Et + 255) / 256, 256, 0, stream>>>(dst, alpha, amax, den, E, Et);
  k_aggregate<<<1024, 256, 0, stream>>>(src, dst, alpha, den, g, part, E, Et);
  k_final    <<<1, HF, 0, stream>>>(part, b2, (float*)d_out, N);
}
